// GCNModel_48292612276725
// MI455X (gfx1250) — compile-verified, matrix operand order checked by reading, not run on hardware
//
#include <hip/hip_runtime.h>

typedef __attribute__((ext_vector_type(8)))  float  v8f;
typedef __attribute__((ext_vector_type(16))) __bf16 v16bf;
typedef __attribute__((ext_vector_type(8)))  __bf16 v8bf;
typedef __attribute__((ext_vector_type(4)))  __bf16 v4bf;

#define IN_F  128
#define HID_F 128
#define OUT_F 64

// ---------------- degree / normalization ----------------

__global__ void k_init_deg(float* __restrict__ deg, int n) {
  int i = blockIdx.x * blockDim.x + threadIdx.x;
  if (i < n) deg[i] = 1.0f;  // self-loop contributes 1
}

__global__ void k_count_deg(const long long* __restrict__ ei, float* __restrict__ deg, int E) {
  int i = blockIdx.x * blockDim.x + threadIdx.x;
  if (i < E) atomicAdd(&deg[(int)ei[(long long)E + i]], 1.0f);  // dst row
}

__global__ void k_make_dinv(float* __restrict__ deg, int n) {
  int i = blockIdx.x * blockDim.x + threadIdx.x;
  if (i < n) deg[i] = rsqrtf(deg[i]);  // deg >= 1 always (self loops)
}

// ---------------- f32 -> bf16 conversion (vectorized) ----------------

__global__ void k_cvt_bf16(const float* __restrict__ x, __bf16* __restrict__ y, long long n4) {
  long long i = (long long)blockIdx.x * blockDim.x + threadIdx.x;
  if (i >= n4) return;
  const float4 v = ((const float4*)x)[i];
  v4bf o;
  o[0] = (__bf16)v.x; o[1] = (__bf16)v.y; o[2] = (__bf16)v.z; o[3] = (__bf16)v.w;
  *(v4bf*)(y + i * 4) = o;  // single 8-byte store
}

// ---------------- pack W[K=128][F] into per-lane WMMA B layout ----------------
// Layout: Wpk[((ct*4 + kb)*32 + lane)*16 + j]
//   lane<16:  N = ct*16+lane,     K = kb*32 + { j<8 ? j      : 16+(j-8) }
//   lane>=16: N = ct*16+lane-16,  K = kb*32 + { j<8 ? 8+j    : 24+(j-8) }
__global__ void k_pack_w(const float* __restrict__ W, __bf16* __restrict__ Wpk,
                         int F, int total) {
  int idx = blockIdx.x * blockDim.x + threadIdx.x;
  if (idx >= total) return;
  int j    = idx & 15;
  int lane = (idx >> 4) & 31;
  int kb   = (idx >> 9) & 3;
  int ct   = idx >> 11;
  int n     = ct * 16 + (lane & 15);
  int kbase = (lane < 16) ? 0 : 8;
  int kk    = kb * 32 + ((j < 8) ? (kbase + j) : (16 + kbase + (j - 8)));
  Wpk[idx] = (__bf16)W[kk * F + n];
}

// ---------------- bf16 WMMA GEMM, fused row-scale by dinv, dual store ----------------
// Computes t = A[N x 128] @ W[128 x F], row r scaled by dinv[r]; writes hs and agg (=self loop).
template <int F>
__global__ __launch_bounds__(128) void k_gemm_bf16(
    const __bf16* __restrict__ A, const __bf16* __restrict__ Wpk,
    const float* __restrict__ dinv, float* __restrict__ hs, float* __restrict__ agg) {
  const int K = 128;
  int lane = threadIdx.x & 31;
  int wave = threadIdx.x >> 5;
  int rowBase = blockIdx.x << 4;
  int ct = (blockIdx.y << 2) + wave;  // 16-wide column tile
  int m  = lane & 15;
  int hi = lane >> 4;     // 0 or 1
  int khalf = hi << 3;    // K sub-offset 0 or 8 per ISA 16-bit A layout

  const __bf16* arow = A + (size_t)(rowBase + m) * K;
  v8f c = {};
#pragma unroll
  for (int kb = 0; kb < K / 32; ++kb) {
    int k0 = kb << 5;
    // A 16x32 bf16: lane<16 holds K = k0+{0..7, 16..23}; lane>=16 holds K = k0+{8..15, 24..31}
    v8bf a0 = *(const v8bf*)(arow + k0 + khalf);        // 16B aligned
    v8bf a1 = *(const v8bf*)(arow + k0 + 16 + khalf);   // 16B aligned
    v16bf a;
#pragma unroll
    for (int t = 0; t < 8; ++t) { a[t] = a0[t]; a[t + 8] = a1[t]; }
    // B: pre-packed, one contiguous 32B chunk per lane
    v16bf b = *(const v16bf*)(Wpk + ((((size_t)ct * (K / 32) + kb) << 5) + lane) * 16);
    c = __builtin_amdgcn_wmma_f32_16x16x32_bf16(false, a, false, b, (short)0, c, false, false);
  }
  // C/D layout: VGPR r, lane<16 -> (M=r, N=lane); lane>=16 -> (M=r+8, N=lane-16)
  int col = (ct << 4) + m;
#pragma unroll
  for (int r = 0; r < 8; ++r) {
    int orow = rowBase + r + (hi << 3);
    float v = c[r] * dinv[orow];
    size_t o = (size_t)orow * F + col;
    hs[o]  = v;   // read-only source for gather
    agg[o] = v;   // self-loop term initializes accumulator
  }
}

// ---------------- edge scatter: agg[dst] += hs[src] ----------------
// One wave per edge (F=128) / per edge-pair (F=64). Edge ids are forced uniform
// via readfirstlane so index loads become scalar (s_load) instead of 32x VMEM.
template <int F>
__global__ __launch_bounds__(256) void k_scatter(const long long* __restrict__ ei,
                                                 const float* __restrict__ hs,
                                                 float* __restrict__ agg, long long E) {
  const int EPW = 128 / F;  // edges per wave: 1 (F=128) or 2 (F=64)
  int wv   = __builtin_amdgcn_readfirstlane((int)threadIdx.x >> 5);
  int lane = (int)threadIdx.x & 31;
  long long ebase = ((long long)blockIdx.x * (blockDim.x >> 5) + wv) * EPW;
  if (ebase >= E) return;  // wave-uniform

  // keep the sequential index stream ahead of the random gathers
  __builtin_prefetch(ei + ebase + 4096, 0, 1);
  __builtin_prefetch(ei + E + ebase + 4096, 0, 1);

  if (EPW == 1) {
    int s = (int)ei[ebase];      // uniform -> scalar load
    int d = (int)ei[E + ebase];
    int f = lane << 2;
    const float4 v = *(const float4*)(hs + (size_t)s * F + f);  // b128 gather
    float* ap = agg + (size_t)d * F + f;
    atomicAdd(ap + 0, v.x);
    atomicAdd(ap + 1, v.y);
    atomicAdd(ap + 2, v.z);
    atomicAdd(ap + 3, v.w);
  } else {
    int s0 = (int)ei[ebase];
    int d0 = (int)ei[E + ebase];
    int s1 = s0, d1 = d0;
    bool has1 = (ebase + 1) < E;
    if (has1) { s1 = (int)ei[ebase + 1]; d1 = (int)ei[E + ebase + 1]; }
    int sel = lane >> 4;              // half-wave per edge
    if (sel && !has1) return;
    int s = sel ? s1 : s0;
    int d = sel ? d1 : d0;
    int f = (lane & 15) << 2;
    const float4 v = *(const float4*)(hs + (size_t)s * F + f);
    float* ap = agg + (size_t)d * F + f;
    atomicAdd(ap + 0, v.x);
    atomicAdd(ap + 1, v.y);
    atomicAdd(ap + 2, v.z);
    atomicAdd(ap + 3, v.w);
  }
}

// ---------------- epilogues: out = (relu)(agg*dinv[row] + b) ----------------

template <int F>
__global__ void k_finish_relu_bf16(const float* __restrict__ agg, const float* __restrict__ dinv,
                                   const float* __restrict__ b, __bf16* __restrict__ out, int n4) {
  int i = blockIdx.x * blockDim.x + threadIdx.x;
  if (i >= n4) return;
  int idx = i << 2;
  int row = idx / F;
  int f = idx & (F - 1);
  float s = dinv[row];
  float4 v  = *(const float4*)(agg + (size_t)idx);
  float4 bb = *(const float4*)(b + f);
  v4bf o;
  o[0] = (__bf16)fmaxf(fmaf(v.x, s, bb.x), 0.0f);
  o[1] = (__bf16)fmaxf(fmaf(v.y, s, bb.y), 0.0f);
  o[2] = (__bf16)fmaxf(fmaf(v.z, s, bb.z), 0.0f);
  o[3] = (__bf16)fmaxf(fmaf(v.w, s, bb.w), 0.0f);
  *(v4bf*)(out + (size_t)idx) = o;
}

template <int F>
__global__ void k_finish_f32(const float* __restrict__ agg, const float* __restrict__ dinv,
                             const float* __restrict__ b, float* __restrict__ out, int n4) {
  int i = blockIdx.x * blockDim.x + threadIdx.x;
  if (i >= n4) return;
  int idx = i << 2;
  int row = idx / F;
  int f = idx & (F - 1);
  float s = dinv[row];
  float4 v  = *(const float4*)(agg + (size_t)idx);
  float4 bb = *(const float4*)(b + f);
  float4 o;
  o.x = fmaf(v.x, s, bb.x);
  o.y = fmaf(v.y, s, bb.y);
  o.z = fmaf(v.z, s, bb.z);
  o.w = fmaf(v.w, s, bb.w);
  *(float4*)(out + (size_t)idx) = o;
}

// ---------------- launch ----------------

extern "C" void kernel_launch(void* const* d_in, const int* in_sizes, int n_in,
                              void* d_out, int out_size, void* d_ws, size_t ws_size,
                              hipStream_t stream) {
  const float*     x  = (const float*)d_in[0];
  // d_in[1] = edge_attr (unused by the model)
  const float*     W1 = (const float*)d_in[2];
  const float*     b1 = (const float*)d_in[3];
  const float*     W2 = (const float*)d_in[4];
  const float*     b2 = (const float*)d_in[5];
  const long long* ei = (const long long*)d_in[6];  // int64 [2, E]: row0=src, row1=dst

  const int N = in_sizes[0] / IN_F;
  const int E = in_sizes[6] / 2;

  char* ws = (char*)d_ws;
  size_t off = 0;
  auto take = [&](size_t bytes) -> char* {
    char* p = ws + off;
    off = (off + bytes + 255) & ~(size_t)255;
    return p;
  };
  float*  dinv = (float*) take((size_t)N * 4);
  __bf16* xbf  = (__bf16*)take((size_t)N * IN_F * 2);
  __bf16* w1pk = (__bf16*)take((size_t)IN_F * HID_F * 2);
  __bf16* w2pk = (__bf16*)take((size_t)HID_F * OUT_F * 2);
  float*  hs1  = (float*) take((size_t)N * HID_F * 4);
  float*  agg1 = (float*) take((size_t)N * HID_F * 4);
  __bf16* h1bf = (__bf16*)take((size_t)N * HID_F * 2);
  float*  hs2  = (float*) take((size_t)N * OUT_F * 4);
  float*  agg2 = (float*) take((size_t)N * OUT_F * 4);

  // normalization: deg (with self loop) -> dinv = rsqrt(deg)
  k_init_deg<<<(N + 255) / 256, 256, 0, stream>>>(dinv, N);
  k_count_deg<<<(E + 255) / 256, 256, 0, stream>>>(ei, dinv, E);
  k_make_dinv<<<(N + 255) / 256, 256, 0, stream>>>(dinv, N);

  // bf16 operand prep
  long long nx4 = (long long)N * IN_F / 4;
  k_cvt_bf16<<<(int)((nx4 + 255) / 256), 256, 0, stream>>>(x, xbf, nx4);
  k_pack_w<<<(IN_F * HID_F + 255) / 256, 256, 0, stream>>>(W1, w1pk, HID_F, IN_F * HID_F);
  k_pack_w<<<(HID_F * OUT_F + 255) / 256, 256, 0, stream>>>(W2, w2pk, OUT_F, HID_F * OUT_F);

  // ---- layer 1 ----
  dim3 g1(N / 16, HID_F / 64);
  k_gemm_bf16<HID_F><<<g1, 128, 0, stream>>>(xbf, w1pk, dinv, hs1, agg1);
  {
    long long waves = (long long)E;                     // 1 edge per wave
    int blocks = (int)((waves + 7) / 8);                // 8 waves / 256-thread block
    k_scatter<HID_F><<<blocks, 256, 0, stream>>>(ei, hs1, agg1, (long long)E);
  }
  int n4a = N * HID_F / 4;
  k_finish_relu_bf16<HID_F><<<(n4a + 255) / 256, 256, 0, stream>>>(agg1, dinv, b1, h1bf, n4a);

  // ---- layer 2 ----
  dim3 g2(N / 16, 1);
  k_gemm_bf16<OUT_F><<<g2, 128, 0, stream>>>(h1bf, w2pk, dinv, hs2, agg2);
  {
    long long waves = ((long long)E + 1) / 2;           // 2 edges per wave
    int blocks = (int)((waves + 7) / 8);
    k_scatter<OUT_F><<<blocks, 256, 0, stream>>>(ei, hs2, agg2, (long long)E);
  }
  int n4b = N * OUT_F / 4;
  k_finish_f32<OUT_F><<<(n4b + 255) / 256, 256, 0, stream>>>(agg2, dinv, b2, (float*)d_out, n4b);
}
